// Temporal_Adapter_78091095376427
// MI455X (gfx1250) — matrix operand order repaired and checked by reading, not compiled, not run on hardware
//
#include <hip/hip_runtime.h>
#include <hip/hip_bf16.h>
#include <math.h>

// ---------------------------------------------------------------------------
// Problem constants (from reference): T_FR=8, N_TOK=257, HW=16, K=3, HEADS=4
//   D=768, Dh=192, B=8  -> M_full = 8*2056 = 16448 rows, M_patch = 16384 tokens
// ---------------------------------------------------------------------------
#define TFR   8
#define NTOK  257
#define BATCH 8
#define DIM   768
#define DH    192
#define DCONV 64
#define HEADS 4
#define HDIM  48
#define M_FULL  (TFR * BATCH * NTOK)    // 16448
#define M_PATCH (BATCH * TFR * 16 * 16) // 16384

typedef __attribute__((ext_vector_type(16))) _Float16 v16h;
typedef __attribute__((ext_vector_type(8)))  float    v8f;
typedef __attribute__((ext_vector_type(4)))  float    f4;
typedef __attribute__((__vector_size__(16))) int      i4v;  // builtin's pointee

enum { EP_NONE = 0, EP_RES = 1, EP_GELU_RES = 2 };

// -------- gfx1250 async global->LDS copy (ASYNCcnt-tracked, no VGPR data) ---
#if __has_builtin(__builtin_amdgcn_global_load_async_to_lds_b128)
#define USE_ASYNC_LDS 1
#endif

__device__ __forceinline__ void async_b128(const float* g, float* l) {
#ifdef USE_ASYNC_LDS
  __builtin_amdgcn_global_load_async_to_lds_b128(
      (__attribute__((address_space(1))) i4v*)g,
      (__attribute__((address_space(3))) i4v*)l, 0, 0);
#else
  *(f4*)l = *(const f4*)g;  // fallback: plain LDS store
#endif
}

__device__ __forceinline__ void wait_async_zero() {
#ifdef USE_ASYNC_LDS
#if __has_builtin(__builtin_amdgcn_s_wait_asynccnt)
  __builtin_amdgcn_s_wait_asynccnt(0);
#else
  asm volatile("s_wait_asynccnt 0" ::: "memory");
#endif
#endif
}

// Read 16 fp32 (two 8-float contiguous runs) from LDS and convert to a packed
// v16h fragment (element j -> VGPR j/2 half j%2, matching CDNA5 16-bit A/B
// VGPR layouts in CDNA5 ISA §7.12.2).
__device__ __forceinline__ v16h load_frag16(const float* __restrict__ p0,
                                            const float* __restrict__ p1) {
  f4 x0 = *(const f4*)(p0);
  f4 x1 = *(const f4*)(p0 + 4);
  f4 x2 = *(const f4*)(p1);
  f4 x3 = *(const f4*)(p1 + 4);
  v16h r;
  r[0]  = (_Float16)x0.x; r[1]  = (_Float16)x0.y;
  r[2]  = (_Float16)x0.z; r[3]  = (_Float16)x0.w;
  r[4]  = (_Float16)x1.x; r[5]  = (_Float16)x1.y;
  r[6]  = (_Float16)x1.z; r[7]  = (_Float16)x1.w;
  r[8]  = (_Float16)x2.x; r[9]  = (_Float16)x2.y;
  r[10] = (_Float16)x2.z; r[11] = (_Float16)x2.w;
  r[12] = (_Float16)x3.x; r[13] = (_Float16)x3.y;
  r[14] = (_Float16)x3.z; r[15] = (_Float16)x3.w;
  return r;
}

#define TSTR 36  // LDS tile row stride in floats (16B-aligned, bank-rotating)

// C[M,N] = A[M,K] @ W[N,K]^T + bias[N]  (+ residual / gelu per EPI)
// Block = 4 waves; block tile 64x64 (each wave: 16 rows x 64 cols, 4 accums).
// A/B 64x32 K-slices staged in LDS via async copies, double-buffered so the
// next slice's DMA overlaps the current slice's WMMA work.
// Requires M%64==0, N%64==0, K%32==0 (true for every GEMM in this model).
template <int EPI>
__global__ __launch_bounds__(128) void gemm_wmma_kernel(
    const float* __restrict__ A, const float* __restrict__ W,
    const float* __restrict__ bias, const float* __restrict__ R,
    float* __restrict__ C, int M, int N, int Kd) {
  __shared__ __align__(16) float sA[2][64 * TSTR];
  __shared__ __align__(16) float sB[2][64 * TSTR];

  const int tid  = threadIdx.x;
  const int lane = tid & 31;
  const int wave = tid >> 5;
  const int mblk = blockIdx.x * 64;
  const int n0   = blockIdx.y * 64;
  const int lm   = lane & 15;  // row (A) / col (B) within 16x16 tile
  const int hi   = lane >> 4;  // k-half selector

  v8f acc[4] = {};
  const int nk = Kd >> 5;

  // Issue K-slice 0 into buffer 0. 128 threads x 4 chunks of 16B per tile.
  {
    const float* Ag = A + (size_t)mblk * Kd;
    const float* Wg = W + (size_t)n0 * Kd;
#pragma unroll
    for (int c = 0; c < 4; ++c) {
      const int chunk = tid + c * 128;          // 0..511
      const int row = chunk >> 3, o = (chunk & 7) * 4;
      async_b128(Ag + (size_t)row * Kd + o, &sA[0][row * TSTR + o]);
      async_b128(Wg + (size_t)row * Kd + o, &sB[0][row * TSTR + o]);
    }
  }

  for (int ik = 0; ik < nk; ++ik) {
    wait_async_zero();   // slice ik (issued last iteration) has landed in LDS
    __syncthreads();

    if (ik + 1 < nk) {   // prefetch slice ik+1 while computing slice ik
      const int kb = (ik + 1) << 5;
      const int nb = (ik + 1) & 1;
      const float* Ag = A + (size_t)mblk * Kd + kb;
      const float* Wg = W + (size_t)n0 * Kd + kb;
#pragma unroll
      for (int c = 0; c < 4; ++c) {
        const int chunk = tid + c * 128;
        const int row = chunk >> 3, o = (chunk & 7) * 4;
        async_b128(Ag + (size_t)row * Kd + o, &sA[nb][row * TSTR + o]);
        async_b128(Wg + (size_t)row * Kd + o, &sB[nb][row * TSTR + o]);
      }
    }

    const int cb = ik & 1;
    // A fragment: lanes 0-15 hold k=[0..8)&[16..24); lanes 16-31 +8.
    const float* ap = &sA[cb][(wave * 16 + lm) * TSTR + hi * 8];
    v16h afrag = load_frag16(ap, ap + 16);
#pragma unroll
    for (int i = 0; i < 4; ++i) {
      // B fragment: lane lm = column n; lanes 0-15 hold k=[0..8)&[16..24); lanes 16-31 +8.
      const float* bp = &sB[cb][(16 * i + lm) * TSTR + hi * 8];
      v16h bfrag = load_frag16(bp, bp + 16);
      acc[i] = __builtin_amdgcn_wmma_f32_16x16x32_f16(
          false, afrag, false, bfrag, (short)0, acc[i], false, false);
    }
    __syncthreads();     // all waves done reading buf cb before it is reissued
  }

  // Epilogue: C/D layout -> VGPR r, lane l : row = m0 + r + 8*(l>>4), col = l&15
  const int m0 = mblk + wave * 16;
#pragma unroll
  for (int i = 0; i < 4; ++i) {
    const int n   = n0 + 16 * i + lm;
    const float bv = bias[n];
#pragma unroll
    for (int r = 0; r < 8; ++r) {
      const int row = m0 + r + 8 * hi;
      const size_t off = (size_t)row * N + n;
      float v = acc[i][r] + bv;
      if (EPI == EP_RES) v += R[off];
      if (EPI == EP_GELU_RES) {
        v += R[off];
        v = 0.5f * v * (1.0f + erff(v * 0.70710678118654752f));  // exact GELU
      }
      C[off] = v;
    }
  }
}

// token id: tok = ((b*8 + t)*16 + h)*16 + w ; source row in (T,BN,C) layout
__device__ __forceinline__ size_t patch_row(int tok) {
  const int w = tok & 15, h = (tok >> 4) & 15, t = (tok >> 8) & 7, b = tok >> 11;
  return (size_t)t * (BATCH * NTOK) + (size_t)b * NTOK + 1 + h * 16 + w;
}

// LayerNorm over C=192 for the 16384 patch tokens; one wave per token.
__global__ __launch_bounds__(256) void ln_kernel(
    const float* __restrict__ S2, const float* __restrict__ g,
    const float* __restrict__ bt, float* __restrict__ out) {
  const int tok  = (blockIdx.x * 256 + threadIdx.x) >> 5;
  const int lane = threadIdx.x & 31;
  if (tok >= M_PATCH) return;
  const float* src = S2 + patch_row(tok) * DH;
  float vals[6], s = 0.f, sq = 0.f;
#pragma unroll
  for (int j = 0; j < 6; ++j) {
    float v = src[lane + 32 * j];
    vals[j] = v; s += v; sq += v * v;
  }
#pragma unroll
  for (int off = 16; off; off >>= 1) {
    s  += __shfl_xor(s,  off, 32);
    sq += __shfl_xor(sq, off, 32);
  }
  const float mu  = s * (1.0f / DH);
  const float var = sq * (1.0f / DH) - mu * mu;
  const float rs  = rsqrtf(var + 1e-5f);
  float* dst = out + (size_t)tok * DH;
#pragma unroll
  for (int j = 0; j < 6; ++j) {
    const int c = lane + 32 * j;
    dst[c] = (vals[j] - mu) * rs * g[c] + bt[c];
  }
}

// NA3D attention: one thread per (token, head). K=3 clipped window -> 27 nbrs.
__global__ __launch_bounds__(256) void na3d_attn_kernel(
    const float* __restrict__ qkv, const float* __restrict__ rpb,
    float* __restrict__ out) {
  const int tid  = blockIdx.x * 256 + threadIdx.x;   // [0, 65536)
  const int head = tid & 3;
  const int tok  = tid >> 2;
  const int w = tok & 15, h = (tok >> 4) & 15, t = (tok >> 8) & 7, b = tok >> 11;
  int st = t - 1; st = st < 0 ? 0 : (st > TFR - 3 ? TFR - 3 : st);
  int sh = h - 1; sh = sh < 0 ? 0 : (sh > 13 ? 13 : sh);
  int sw = w - 1; sw = sw < 0 ? 0 : (sw > 13 ? 13 : sw);

  const float* qp = qkv + (size_t)tok * (3 * DH) + head * HDIM;
  float q[HDIM];
#pragma unroll
  for (int c = 0; c < HDIM; ++c) q[c] = qp[c] * 0.14433756729740643f; // 48^-1/2

  float sc[27];
  float mx = -1e30f;
  int j = 0;
#pragma unroll
  for (int dt = 0; dt < 3; ++dt)
#pragma unroll
    for (int dh = 0; dh < 3; ++dh)
#pragma unroll
      for (int dw = 0; dw < 3; ++dw, ++j) {
        const int nt = st + dt, nh = sh + dh, nw = sw + dw;
        const int ntok = ((b * 8 + nt) * 16 + nh) * 16 + nw;
        const float* kp = qkv + (size_t)ntok * (3 * DH) + DH + head * HDIM;
        float d = 0.f;
#pragma unroll
        for (int c = 0; c < HDIM; ++c) d += q[c] * kp[c];
        d += rpb[((head * 5 + (nt - t + 2)) * 5 + (nh - h + 2)) * 5 + (nw - w + 2)];
        sc[j] = d;
        mx = fmaxf(mx, d);
      }
  float sum = 0.f;
#pragma unroll
  for (j = 0; j < 27; ++j) { sc[j] = __expf(sc[j] - mx); sum += sc[j]; }
  const float inv = 1.0f / sum;

  float o[HDIM];
#pragma unroll
  for (int c = 0; c < HDIM; ++c) o[c] = 0.f;
  j = 0;
#pragma unroll
  for (int dt = 0; dt < 3; ++dt)
#pragma unroll
    for (int dh = 0; dh < 3; ++dh)
#pragma unroll
      for (int dw = 0; dw < 3; ++dw, ++j) {
        const int nt = st + dt, nh = sh + dh, nw = sw + dw;
        const int ntok = ((b * 8 + nt) * 16 + nh) * 16 + nw;
        const float* vp = qkv + (size_t)ntok * (3 * DH) + 2 * DH + head * HDIM;
        const float wj = sc[j] * inv;
#pragma unroll
        for (int c = 0; c < HDIM; ++c) o[c] += wj * vp[c];
      }
  float* op = out + (size_t)tok * DH + head * HDIM;
#pragma unroll
  for (int c = 0; c < HDIM; ++c) op[c] = o[c];
}

// S2[(t,bn),c] += proj[tok,c] for patch tokens (residual around NA3D).
__global__ __launch_bounds__(256) void scatter_add_kernel(
    const float* __restrict__ proj, float* __restrict__ S2) {
  const int idx = blockIdx.x * 256 + threadIdx.x;  // 16384*192
  const int tok = idx / DH;
  const int c   = idx - tok * DH;
  S2[patch_row(tok) * DH + c] += proj[idx];
}

extern "C" void kernel_launch(void* const* d_in, const int* in_sizes, int n_in,
                              void* d_out, int out_size, void* d_ws, size_t ws_size,
                              hipStream_t stream) {
  const float* x       = (const float*)d_in[0];   // (8, 2056, 768)
  const float* fc1_w   = (const float*)d_in[1];   // (192, 768)
  const float* fc1_b   = (const float*)d_in[2];
  const float* convA_w = (const float*)d_in[3];   // (64, 192)
  const float* convA_b = (const float*)d_in[4];
  const float* convB_w = (const float*)d_in[5];   // (192, 64)
  const float* convB_b = (const float*)d_in[6];
  const float* ln_g    = (const float*)d_in[7];
  const float* ln_b    = (const float*)d_in[8];
  const float* qkv_w   = (const float*)d_in[9];   // (576, 192)
  const float* qkv_b   = (const float*)d_in[10];
  const float* rpb     = (const float*)d_in[11];  // (4, 5, 5, 5)
  const float* proj_w  = (const float*)d_in[12];  // (192, 192)
  const float* proj_b  = (const float*)d_in[13];
  const float* fc2_w   = (const float*)d_in[14];  // (768, 192)
  const float* fc2_b   = (const float*)d_in[15];
  float* out = (float*)d_out;

  // Workspace layout (floats), with region reuse:
  float* S1  = (float*)d_ws;                 // M_FULL*192  (later: attn out)
  float* Hc  = S1 + (size_t)M_FULL * DH;     // M_FULL*64
  float* S2  = Hc + (size_t)M_FULL * DCONV;  // M_FULL*192  (persists, updated)
  float* LNB = S2 + (size_t)M_FULL * DH;     // M_PATCH*192 (later: proj out)
  float* QKV = LNB + (size_t)M_PATCH * DH;   // M_PATCH*576
  float* ATT = S1;    // S1 free after S2 computed
  float* PRJ = LNB;   // LN free after qkv GEMM

  dim3 gblk(128);

  // 1) S1 = x @ fc1_w^T + fc1_b
  gemm_wmma_kernel<EP_NONE><<<dim3(M_FULL / 64, DH / 64), gblk, 0, stream>>>(
      x, fc1_w, fc1_b, nullptr, S1, M_FULL, DH, DIM);
  // 2) Hc = S1 @ convA_w^T + convA_b
  gemm_wmma_kernel<EP_NONE><<<dim3(M_FULL / 64, DCONV / 64), gblk, 0, stream>>>(
      S1, convA_w, convA_b, nullptr, Hc, M_FULL, DCONV, DH);
  // 3) S2 = gelu(S1 + Hc @ convB_w^T + convB_b)
  gemm_wmma_kernel<EP_GELU_RES><<<dim3(M_FULL / 64, DH / 64), gblk, 0, stream>>>(
      Hc, convB_w, convB_b, S1, S2, M_FULL, DH, DCONV);
  // 4) LN over patch tokens
  ln_kernel<<<dim3(M_PATCH / 8), dim3(256), 0, stream>>>(S2, ln_g, ln_b, LNB);
  // 5) QKV = LN @ qkv_w^T + qkv_b
  gemm_wmma_kernel<EP_NONE><<<dim3(M_PATCH / 64, (3 * DH) / 64), gblk, 0, stream>>>(
      LNB, qkv_w, qkv_b, nullptr, QKV, M_PATCH, 3 * DH, DH);
  // 6) NA3D attention (per token/head)
  na3d_attn_kernel<<<dim3((M_PATCH * HEADS) / 256), dim3(256), 0, stream>>>(
      QKV, rpb, ATT);
  // 7) PRJ = ATT @ proj_w^T + proj_b
  gemm_wmma_kernel<EP_NONE><<<dim3(M_PATCH / 64, DH / 64), gblk, 0, stream>>>(
      ATT, proj_w, proj_b, nullptr, PRJ, M_PATCH, DH, DH);
  // 8) S2[patch rows] += PRJ   (residual around attention)
  scatter_add_kernel<<<dim3((M_PATCH * DH) / 256), dim3(256), 0, stream>>>(PRJ, S2);
  // 9) out = x + S2 @ fc2_w^T + fc2_b
  gemm_wmma_kernel<EP_RES><<<dim3(M_FULL / 64, DIM / 64), gblk, 0, stream>>>(
      S2, fc2_w, fc2_b, x, out, M_FULL, DIM, DH);
}
